// NTXent_19774029431199
// MI455X (gfx1250) — matrix operand order repaired and checked by reading, not compiled, not run on hardware
//
#include <hip/hip_runtime.h>
#include <hip/hip_bf16.h>

typedef __attribute__((ext_vector_type(16))) __bf16 v16bf;
typedef __attribute__((ext_vector_type(8)))  float  v8f;

#define DIMK   512
#define BATCH  4096
#define B2     8192            // 2*batch rows
#define KCH    (DIMK / 32)     // 16 k-chunks of 32
#define ROWS_PER_WG 128
#define SPLIT  8
#define JT_TOTAL (B2 / 16)           // 512 column tiles
#define JT_PER_WG (JT_TOTAL / SPLIT) // 64 per workgroup

#if defined(__HIP_DEVICE_COMPILE__) && __has_builtin(__builtin_amdgcn_global_load_async_to_lds_b128)
#define HAS_ASYNC_LDS 1
typedef int v4i __attribute__((vector_size(16)));       // 128-bit payload type
typedef __attribute__((address_space(1))) v4i g_v4i;    // global
typedef __attribute__((address_space(3))) v4i l_v4i;    // LDS
#else
#define HAS_ASYNC_LDS 0
#endif

// ---------------------------------------------------------------------------
// Kernel 0: zero the per-row exp-sum accumulators (atomics target).
// ---------------------------------------------------------------------------
__global__ void ntxent_zero(float* __restrict__ p, int n) {
    int i = blockIdx.x * 256 + threadIdx.x;
    if (i < n) p[i] = 0.0f;
}

// ---------------------------------------------------------------------------
// Kernel 1: L2-normalize rows (f32), emit bf16 reps [8192][512], and the
// f32 positives pos[r] = z_i[r] . z_j[r] (kept full precision).
// ---------------------------------------------------------------------------
__global__ __launch_bounds__(256) void ntxent_normalize(
    const float* __restrict__ l1, const float* __restrict__ l2,
    __bf16* __restrict__ R, float* __restrict__ pos) {
    const int r = blockIdx.x;
    const int t = threadIdx.x;
    const float* a = l1 + (size_t)r * DIMK;
    const float* b = l2 + (size_t)r * DIMK;
    float a0 = a[t], a1 = a[t + 256];
    float b0 = b[t], b1 = b[t + 256];
    __shared__ float s1[256], s2[256], s3[256];
    s1[t] = a0 * a0 + a1 * a1;
    s2[t] = b0 * b0 + b1 * b1;
    s3[t] = a0 * b0 + a1 * b1;
    __syncthreads();
    for (int off = 128; off > 0; off >>= 1) {
        if (t < off) {
            s1[t] += s1[t + off];
            s2[t] += s2[t + off];
            s3[t] += s3[t + off];
        }
        __syncthreads();
    }
    const float inv1 = 1.0f / sqrtf(s1[0]);
    const float inv2 = 1.0f / sqrtf(s2[0]);
    __bf16* r1 = R + (size_t)r * DIMK;
    __bf16* r2 = R + (size_t)(r + BATCH) * DIMK;
    r1[t]       = (__bf16)(a0 * inv1);
    r1[t + 256] = (__bf16)(a1 * inv1);
    r2[t]       = (__bf16)(b0 * inv2);
    r2[t + 256] = (__bf16)(b1 * inv2);
    if (t == 0) pos[r] = s3[0] * inv1 * inv2;
}

// ---------------------------------------------------------------------------
// Kernel 2: fused sim = R R^T (bf16 WMMA, f32 acc) -> exp(sim/T) -> row sums
// with diagonal masked. Grid: (64 row-blocks, 8 column splits), 256 threads.
// Wave w owns rows [bx*128 + w*16, +16): A panel held in VGPRs for all jt.
// B tiles double-buffered through LDS via CDNA5 async global->LDS DMA.
// ---------------------------------------------------------------------------
__global__ __launch_bounds__(256) void ntxent_simsum(
    const __bf16* __restrict__ R, float* __restrict__ rowsum) {
    // 2 buffers x 16 rows x (64+1 pad) uint4  -> 33,280 bytes LDS
    __shared__ uint4 ldsB[2][16 * 65];

    const int tid  = threadIdx.x;
    const int wave = tid >> 5;
    const int lane = tid & 31;
    const bool hi  = lane >= 16;
    const int l16  = lane & 15;

    const int rowBase = blockIdx.x * ROWS_PER_WG + wave * 16;
    const int jt0     = blockIdx.y * JT_PER_WG;

    const uint4* Rq = (const uint4*)R;  // 64 uint4 (=512 bf16) per row

    union V16 { v16bf v; uint4 q[2]; };

    // ---- A panel (16 rows x 512) resident in registers, WMMA A layout:
    // lane<16: halves = K[k0..k0+7], K[k0+16..k0+23]  (row = rowBase+lane)
    // lane>=16: halves = K[k0+8..k0+15], K[k0+24..k0+31]
    const int arow = rowBase + l16;
    const uint4* Arow = Rq + (size_t)arow * 64;
    V16 areg[KCH];
#pragma unroll
    for (int kc = 0; kc < KCH; ++kc) {
        const int q0 = kc * 4 + (hi ? 1 : 0);
        areg[kc].q[0] = Arow[q0];
        areg[kc].q[1] = Arow[q0 + 2];
    }

    float acc[8];
#pragma unroll
    for (int v = 0; v < 8; ++v) acc[v] = 0.0f;

    // 16 WMMAs on buffer (j&1), then exp epilogue with diagonal masking.
    auto computeTile = [&](int j) {
        const int jt = jt0 + j;
        const uint4* bb = ldsB[j & 1];
        v8f d = {};
#pragma unroll
        for (int kc = 0; kc < KCH; ++kc) {
            V16 bv;
            const int qi = l16 * 65 + kc * 4 + (hi ? 1 : 0);
            bv.q[0] = bb[qi];
            bv.q[1] = bb[qi + 2];
            d = __builtin_amdgcn_wmma_f32_16x16x32_bf16(
                    false, areg[kc].v, false, bv.v, (short)0, d, false, false);
        }
        const int col   = jt * 16 + l16;
        const int rbase = rowBase + (hi ? 8 : 0);
#pragma unroll
        for (int v = 0; v < 8; ++v) {
            const float e = __expf(2.0f * d[v]);   // 1/TEMPERATURE = 2
            if (rbase + v != col) acc[v] += e;
        }
    };

#if HAS_ASYNC_LDS
    // ---- CDNA5 async copy path: global -> LDS DMA, tracked on ASYNCcnt. ----
    auto issueB = [&](int jt, int buf) {
        const uint4* src = Rq + (size_t)(jt * 16) * 64;
#pragma unroll
        for (int i = 0; i < 4; ++i) {
            const int q = tid + 256 * i;
            const int row = q >> 6, c = q & 63;
            __builtin_amdgcn_global_load_async_to_lds_b128(
                (g_v4i*)(src + row * 64 + c),
                (l_v4i*)&ldsB[buf][row * 65 + c],
                0, 0);
        }
    };
    issueB(jt0, 0);
    for (int j = 0; j < JT_PER_WG; ++j) {
#if __has_builtin(__builtin_amdgcn_s_wait_asynccnt)
        __builtin_amdgcn_s_wait_asynccnt(0);
#else
        asm volatile("s_wait_asynccnt 0x0" ::: "memory");
#endif
        __syncthreads();                    // everyone's async fill of buf(j&1) done
        if (j + 1 < JT_PER_WG) issueB(jt0 + j + 1, (j + 1) & 1);
        computeTile(j);
    }
#else
    // ---- Fallback: two-stage VGPR prefetch so loadcnt wait lands one tile
    // after the loads were issued (hidden behind the previous tile's WMMAs).
    uint4 pre[4];
    auto loadG = [&](int jt) {
        const uint4* src = Rq + (size_t)(jt * 16) * 64;
#pragma unroll
        for (int i = 0; i < 4; ++i) {
            const int q = tid + 256 * i;
            pre[i] = src[(q >> 6) * 64 + (q & 63)];
        }
    };
    auto storeL = [&](int buf) {
        uint4* dst = ldsB[buf];
#pragma unroll
        for (int i = 0; i < 4; ++i) {
            const int q = tid + 256 * i;
            dst[(q >> 6) * 65 + (q & 63)] = pre[i];
        }
    };
    loadG(jt0);
    for (int j = 0; j < JT_PER_WG; ++j) {
        storeL(j & 1);                      // waits on loads issued last iter
        __syncthreads();
        if (j + 1 < JT_PER_WG) loadG(jt0 + j + 1);   // in flight during WMMAs
        computeTile(j);
    }
#endif

    // Reduce across the 16 lanes of each half-wave (columns of the tile).
#pragma unroll
    for (int v = 0; v < 8; ++v) {
        float x = acc[v];
        for (int m = 8; m >= 1; m >>= 1) x += __shfl_xor(x, m, 16);
        if (l16 == 0) atomicAdd(&rowsum[rowBase + (hi ? 8 : 0) + v], x);
    }
}

// ---------------------------------------------------------------------------
// Kernel 3: loss = sum_n [log(denom_n) - pos_{n mod 4096}/T] / (2*4096)
// ---------------------------------------------------------------------------
__global__ __launch_bounds__(256) void ntxent_loss(
    const float* __restrict__ rowsum, const float* __restrict__ pos,
    float* __restrict__ out) {
    const int t = threadIdx.x;
    float s = 0.0f;
    for (int n = t; n < B2; n += 256)
        s += __logf(rowsum[n]) - 2.0f * pos[n & (BATCH - 1)];
    __shared__ float sh[256];
    sh[t] = s;
    __syncthreads();
    for (int off = 128; off > 0; off >>= 1) {
        if (t < off) sh[t] += sh[t + off];
        __syncthreads();
    }
    if (t == 0) out[0] = sh[0] * (1.0f / (2.0f * BATCH));
}

// ---------------------------------------------------------------------------
extern "C" void kernel_launch(void* const* d_in, const int* in_sizes, int n_in,
                              void* d_out, int out_size, void* d_ws, size_t ws_size,
                              hipStream_t stream) {
    const float* l1 = (const float*)d_in[0];
    const float* l2 = (const float*)d_in[1];
    char* ws = (char*)d_ws;

    __bf16* R      = (__bf16*)ws;                              // 8 MB bf16 reps
    float*  pos    = (float*)(ws + (size_t)B2 * DIMK * 2);     // 16 KB
    float*  rowsum = pos + BATCH;                              // 32 KB
    float*  out    = (float*)d_out;

    ntxent_zero<<<(B2 + 255) / 256, 256, 0, stream>>>(rowsum, B2);
    ntxent_normalize<<<BATCH, 256, 0, stream>>>(l1, l2, R, pos);
    dim3 grid(B2 / ROWS_PER_WG, SPLIT);                        // 64 x 8 = 512 WGs
    ntxent_simsum<<<grid, 256, 0, stream>>>(R, rowsum);
    ntxent_loss<<<1, 256, 0, stream>>>(rowsum, pos, out);
}
